// CogTauBlock_420906795751
// MI455X (gfx1250) — matrix operand order repaired and checked by reading, not compiled, not run on hardware
//
#include <hip/hip_runtime.h>
#include <math.h>

#define DIMC  3072
#define HEADS 48
#define HD    64
#define TEMBC 512
#define BATCH 2
#define TTXT  226
#define SVID  1024
#define SEQ   (TTXT + SVID)       // 1250
#define MROWS (BATCH * SEQ)       // 2500
#define FFC   (4 * DIMC)          // 12288

typedef __attribute__((ext_vector_type(16))) __bf16          bf16x16;
typedef __attribute__((ext_vector_type(8)))  float           floatx8;
typedef __attribute__((ext_vector_type(4)))  float           floatx4;
typedef __attribute__((ext_vector_type(8)))  unsigned short  ushortx8;
typedef __attribute__((ext_vector_type(16))) unsigned short  ushortx16;

union Frag {
  ushortx16 u;
  bf16x16   b;
  struct { ushortx8 lo, hi; } h;
};

__device__ __forceinline__ unsigned short f2bf(float f) {
  unsigned int u = __float_as_uint(f);
  u += 0x7fffu + ((u >> 16) & 1u);
  return (unsigned short)(u >> 16);
}

// ---------------------------------------------------------------- small ops
__global__ void silu_k(const float* __restrict__ t, float* __restrict__ o) {
  int i = blockIdx.x * blockDim.x + threadIdx.x;
  if (i < BATCH * TEMBC) { float v = t[i]; o[i] = v / (1.0f + __expf(-v)); }
}

__global__ __launch_bounds__(256)
void mods_k(const float* __restrict__ st, const float* __restrict__ w,
            const float* __restrict__ bias, float* __restrict__ out) {
  int i = blockIdx.x * 256 + threadIdx.x;
  if (i >= BATCH * 6 * DIMC) return;
  int b = i / (6 * DIMC), n = i - b * 6 * DIMC;
  const float* wr = w + (size_t)n * TEMBC;
  const float* tv = st + (size_t)b * TEMBC;
  float acc = bias[n];
  for (int k = 0; k < TEMBC; ++k) acc += tv[k] * wr[k];
  out[i] = acc;
}

// LayerNorm + adaLN modulation -> bf16 "cat" activation
__global__ __launch_bounds__(256)
void lnmod_k(const float* __restrict__ src_e, const float* __restrict__ src_h,
             int cat_layout, const float* __restrict__ g, const float* __restrict__ bb,
             const float* __restrict__ mods, unsigned short* __restrict__ out) {
  __shared__ float red[256];
  __shared__ float stats[2];
  int row = blockIdx.x;
  int b = row / SEQ, s = row - b * SEQ;
  const float* x;
  if (cat_layout) x = src_e + (size_t)row * DIMC;
  else x = (s < TTXT) ? src_e + ((size_t)b * TTXT + s) * DIMC
                      : src_h + ((size_t)b * SVID + (s - TTXT)) * DIMC;
  int tid = threadIdx.x;
  float sum = 0.f, sq = 0.f;
  for (int c = tid; c < DIMC; c += 256) { float v = x[c]; sum += v; sq += v * v; }
  red[tid] = sum; __syncthreads();
  for (int st2 = 128; st2 > 0; st2 >>= 1) { if (tid < st2) red[tid] += red[tid + st2]; __syncthreads(); }
  if (tid == 0) stats[0] = red[0] * (1.0f / DIMC);
  __syncthreads();
  red[tid] = sq; __syncthreads();
  for (int st2 = 128; st2 > 0; st2 >>= 1) { if (tid < st2) red[tid] += red[tid + st2]; __syncthreads(); }
  if (tid == 0) {
    float m = stats[0];
    stats[1] = rsqrtf(red[0] * (1.0f / DIMC) - m * m + 1e-5f);
  }
  __syncthreads();
  float mean = stats[0], inv = stats[1];
  const float* shp = mods + (size_t)b * 6 * DIMC + (s < TTXT ? 3 * DIMC : 0);
  const float* scp = mods + (size_t)b * 6 * DIMC + (s < TTXT ? 4 * DIMC : DIMC);
  for (int c = tid; c < DIMC; c += 256) {
    float nx = (x[c] - mean) * inv * g[c] + bb[c];
    out[(size_t)row * DIMC + c] = f2bf(nx * (1.0f + scp[c]) + shp[c]);
  }
}

// per-head q/k layernorm + rope (video tokens) + v convert, fp32 -> bf16
__global__ __launch_bounds__(256)
void qkrope_k(const float* __restrict__ qf, const float* __restrict__ kf,
              const float* __restrict__ vf,
              const float* __restrict__ nqg, const float* __restrict__ nqb,
              const float* __restrict__ nkg, const float* __restrict__ nkb,
              const float* __restrict__ rc, const float* __restrict__ rs,
              unsigned short* __restrict__ qo, unsigned short* __restrict__ ko,
              unsigned short* __restrict__ vo) {
  int t = blockIdx.x * 256 + threadIdx.x;
  if (t >= BATCH * SEQ * HEADS) return;
  int h = t % HEADS;
  int s = (t / HEADS) % SEQ;
  int b = t / (HEADS * SEQ);
  size_t off = ((size_t)(b * SEQ + s)) * DIMC + (size_t)h * HD;
  bool vid = (s >= TTXT);
  int vp = s - TTXT;
  for (int which = 0; which < 2; ++which) {
    const float* x = (which ? kf : qf) + off;
    const float* gg = which ? nkg : nqg;
    const float* b2 = which ? nkb : nqb;
    unsigned short* o = (which ? ko : qo) + off;
    float sum = 0.f, sq = 0.f;
    for (int d = 0; d < HD; ++d) { float v = x[d]; sum += v; sq += v * v; }
    float mean = sum * (1.0f / HD);
    float inv = rsqrtf(sq * (1.0f / HD) - mean * mean + 1e-6f);
    for (int d = 0; d < HD; d += 2) {
      float a0 = (x[d] - mean) * inv * gg[d] + b2[d];
      float a1 = (x[d + 1] - mean) * inv * gg[d + 1] + b2[d + 1];
      float r0 = a0, r1 = a1;
      if (vid) {
        float c0 = rc[vp * HD + d], c1 = rc[vp * HD + d + 1];
        float s0 = rs[vp * HD + d], s1 = rs[vp * HD + d + 1];
        r0 = a0 * c0 - a1 * s0;
        r1 = a1 * c1 + a0 * s1;
      }
      o[d] = f2bf(r0); o[d + 1] = f2bf(r1);
    }
  }
  const float* xv = vf + off;
  unsigned short* ov = vo + off;
  for (int d = 0; d < HD; ++d) ov[d] = f2bf(xv[d]);
}

// ---------------------------------------------------------------- WMMA GEMM
// C[M,N] = A[M,K](bf16) @ W[N,K](f32, converted via double-buffered LDS)^T.
// 8 waves/block; wave tile = 32 rows x 64 cols (8 WMMAs per 32-k step).
struct GemmP {
  const unsigned short* A;
  const float* W;
  const float* bias;
  float* outF;
  unsigned short* outB;
  int M, N, K;
  const float* mods;
  const float* res_e;
  const float* res_h;
  float* outH;
  float* outE;
};

template <int EPI>
__global__ __launch_bounds__(256)
void gemm_wmma_k(GemmP p) {
  __shared__ alignas(16) unsigned short wtile[2][64 * 32];
  const int tid = threadIdx.x;
  const int wave = tid >> 5;
  const int lane = tid & 31;
  const int lh = lane & 15;
  const int hiw = lane >> 4;
  const int c0 = blockIdx.x * 64;
  const int m0 = blockIdx.y * 256 + wave * 32;
  int arow0 = m0 + lh;      if (arow0 >= p.M) arow0 = p.M - 1;
  int arow1 = m0 + 16 + lh; if (arow1 >= p.M) arow1 = p.M - 1;
  const unsigned short* Arow0 = p.A + (size_t)arow0 * p.K;
  const unsigned short* Arow1 = p.A + (size_t)arow1 * p.K;
  floatx8 acc[2][4];
  #pragma unroll
  for (int mi = 0; mi < 2; ++mi)
    #pragma unroll
    for (int nt = 0; nt < 4; ++nt)
      acc[mi][nt] = (floatx8){0.f,0.f,0.f,0.f,0.f,0.f,0.f,0.f};

  const int e0 = tid * 8;                       // 0..2047, covers 64x32 tile
  const int nl = e0 >> 5;
  const int kk = e0 & 31;
  const float* wrow = p.W + (size_t)(c0 + nl) * p.K + kk;

  // stage W[k..k+32) tile for this block's 64 cols, fp32 -> bf16, into buf
  auto stage = [&](int k, int buf) {
    const float* wp = wrow + k;
    floatx4 w0 = *(const floatx4*)wp;
    floatx4 w1 = *(const floatx4*)(wp + 4);
    ushortx8 wsr;
    wsr[0] = f2bf(w0[0]); wsr[1] = f2bf(w0[1]); wsr[2] = f2bf(w0[2]); wsr[3] = f2bf(w0[3]);
    wsr[4] = f2bf(w1[0]); wsr[5] = f2bf(w1[1]); wsr[6] = f2bf(w1[2]); wsr[7] = f2bf(w1[3]);
    *(ushortx8*)(wtile[buf] + e0) = wsr;
    if (k + 96 <= p.K) __builtin_prefetch((const void*)(wp + 64), 0, 1);
  };

  stage(0, 0);
  int it = 0;
  for (int k = 0; k < p.K; k += 32, ++it) {
    __syncthreads();
    if (k + 32 < p.K) stage(k + 32, (it + 1) & 1);
    const unsigned short* wt = wtile[it & 1];
    Frag a0, a1;
    a0.h.lo = *(const ushortx8*)(Arow0 + k + hiw * 8);
    a0.h.hi = *(const ushortx8*)(Arow0 + k + 16 + hiw * 8);
    a1.h.lo = *(const ushortx8*)(Arow1 + k + hiw * 8);
    a1.h.hi = *(const ushortx8*)(Arow1 + k + 16 + hiw * 8);
    #pragma unroll
    for (int nt = 0; nt < 4; ++nt) {
      Frag bfr;
      const unsigned short* bp = wt + (nt * 16 + lh) * 32 + hiw * 8;
      bfr.h.lo = *(const ushortx8*)bp;
      bfr.h.hi = *(const ushortx8*)(bp + 16);
      acc[0][nt] = __builtin_amdgcn_wmma_f32_16x16x32_bf16(
          false, a0.b, false, bfr.b, (short)0, acc[0][nt], false, false);
      acc[1][nt] = __builtin_amdgcn_wmma_f32_16x16x32_bf16(
          false, a1.b, false, bfr.b, (short)0, acc[1][nt], false, false);
    }
  }

  #pragma unroll
  for (int mi = 0; mi < 2; ++mi) {
    #pragma unroll
    for (int nt = 0; nt < 4; ++nt) {
      int col = c0 + nt * 16 + lh;
      float bv = (EPI == 0) ? 0.0f : p.bias[col];
      #pragma unroll
      for (int r = 0; r < 8; ++r) {
        int row = m0 + mi * 16 + r + hiw * 8;
        if (row >= p.M) continue;
        float val = acc[mi][nt][r] + bv;
        if (EPI == 0) {
          p.outF[(size_t)row * p.N + col] = val;
        } else if (EPI == 1) {            // O-proj: gated residual -> resid1 (cat layout)
          int b = row / SEQ, s = row - b * SEQ;
          float gv, res;
          if (s < TTXT) {
            gv = p.mods[(size_t)b * 6 * DIMC + 5 * DIMC + col];
            res = p.res_e[((size_t)b * TTXT + s) * DIMC + col];
          } else {
            gv = p.mods[(size_t)b * 6 * DIMC + 2 * DIMC + col];
            res = p.res_h[((size_t)b * SVID + (s - TTXT)) * DIMC + col];
          }
          p.outF[(size_t)row * DIMC + col] = res + gv * val;
        } else if (EPI == 2) {            // FF1: gelu(tanh) -> bf16
          float x = val;
          float t = tanhf(0.7978845608028654f * (x + 0.044715f * x * x * x));
          p.outB[(size_t)row * p.N + col] = f2bf(0.5f * x * (1.0f + t));
        } else {                          // FF2: gated residual, split h/e into d_out
          int b = row / SEQ, s = row - b * SEQ;
          float res = p.res_e[(size_t)row * DIMC + col];   // resid1, cat layout
          if (s < TTXT) {
            float gv = p.mods[(size_t)b * 6 * DIMC + 5 * DIMC + col];
            p.outE[((size_t)b * TTXT + s) * DIMC + col] = res + gv * val;
          } else {
            float gv = p.mods[(size_t)b * 6 * DIMC + 2 * DIMC + col];
            p.outH[((size_t)b * SVID + (s - TTXT)) * DIMC + col] = res + gv * val;
          }
        }
      }
    }
  }
}

// ---------------------------------------------------------------- attention
// Flash attention. 4 waves/block cover 64 queries of one (b,h); the 32x64 V^T
// tile is staged into LDS once per block and shared by all 4 waves.
__global__ __launch_bounds__(128)
void attn_k(const unsigned short* __restrict__ Q, const unsigned short* __restrict__ K,
            const unsigned short* __restrict__ V, unsigned short* __restrict__ O) {
  __shared__ alignas(16) unsigned short plds[4][16 * 32];
  __shared__ alignas(16) unsigned short vt[64 * 32];
  const int tid = threadIdx.x;
  const int wave = tid >> 5;
  const int lane = tid & 31;
  const int lh = lane & 15;
  const int hiw = lane >> 4;
  const int q0 = blockIdx.x * 64 + wave * 16;
  const int h = blockIdx.y;
  const int b = blockIdx.z;
  const size_t base = ((size_t)b * SEQ) * DIMC + (size_t)h * HD;

  int qrow = q0 + lh; if (qrow >= SEQ) qrow = SEQ - 1;
  Frag aq0, aq1;
  {
    const unsigned short* qp = Q + base + (size_t)qrow * DIMC;
    aq0.h.lo = *(const ushortx8*)(qp + hiw * 8);
    aq0.h.hi = *(const ushortx8*)(qp + 16 + hiw * 8);
    aq1.h.lo = *(const ushortx8*)(qp + 32 + hiw * 8);
    aq1.h.hi = *(const ushortx8*)(qp + 48 + hiw * 8);
  }
  floatx8 oo[4];
  #pragma unroll
  for (int i = 0; i < 4; ++i) oo[i] = (floatx8){0.f,0.f,0.f,0.f,0.f,0.f,0.f,0.f};
  float mrun[8], lrun[8];
  #pragma unroll
  for (int r = 0; r < 8; ++r) { mrun[r] = -1e30f; lrun[r] = 0.f; }
  const float scale = 0.125f;   // 1/sqrt(64)

  for (int kb = 0; kb < SEQ; kb += 32) {
    floatx8 sc[2];
    #pragma unroll
    for (int kt = 0; kt < 2; ++kt) {
      sc[kt] = (floatx8){0.f,0.f,0.f,0.f,0.f,0.f,0.f,0.f};
      int krow = kb + kt * 16 + lh; if (krow >= SEQ) krow = SEQ - 1;
      const unsigned short* kp = K + base + (size_t)krow * DIMC;
      Frag b0, b1;
      b0.h.lo = *(const ushortx8*)(kp + hiw * 8);
      b0.h.hi = *(const ushortx8*)(kp + 16 + hiw * 8);
      b1.h.lo = *(const ushortx8*)(kp + 32 + hiw * 8);
      b1.h.hi = *(const ushortx8*)(kp + 48 + hiw * 8);
      sc[kt] = __builtin_amdgcn_wmma_f32_16x16x32_bf16(false, aq0.b, false, b0.b, (short)0, sc[kt], false, false);
      sc[kt] = __builtin_amdgcn_wmma_f32_16x16x32_bf16(false, aq1.b, false, b1.b, (short)0, sc[kt], false, false);
    }
    #pragma unroll
    for (int kt = 0; kt < 2; ++kt) {
      bool oob = (kb + kt * 16 + lh) >= SEQ;
      #pragma unroll
      for (int r = 0; r < 8; ++r) sc[kt][r] = oob ? -1e30f : sc[kt][r] * scale;
    }
    #pragma unroll
    for (int r = 0; r < 8; ++r) {
      float v = fmaxf(sc[0][r], sc[1][r]);
      v = fmaxf(v, __shfl_xor(v, 1, 32));
      v = fmaxf(v, __shfl_xor(v, 2, 32));
      v = fmaxf(v, __shfl_xor(v, 4, 32));
      v = fmaxf(v, __shfl_xor(v, 8, 32));
      float mnew = fmaxf(mrun[r], v);
      float f = __expf(mrun[r] - mnew);
      mrun[r] = mnew;
      lrun[r] *= f;
      oo[0][r] *= f; oo[1][r] *= f; oo[2][r] *= f; oo[3][r] *= f;
      float p0 = __expf(sc[0][r] - mnew);
      float p1 = __expf(sc[1][r] - mnew);
      float ps = p0 + p1;
      ps += __shfl_xor(ps, 1, 32);
      ps += __shfl_xor(ps, 2, 32);
      ps += __shfl_xor(ps, 4, 32);
      ps += __shfl_xor(ps, 8, 32);
      lrun[r] += ps;
      plds[wave][(r + hiw * 8) * 32 + lh]      = f2bf(p0);
      plds[wave][(r + hiw * 8) * 32 + 16 + lh] = f2bf(p1);
    }
    {  // cooperative V^T stage: 128 threads, 16 halves each -> vt[d][j]
      int j = tid & 31;
      int dbase = (tid >> 5) * 16;
      int krow = kb + j; if (krow >= SEQ) krow = SEQ - 1;
      const unsigned short* vp = V + base + (size_t)krow * DIMC + dbase;
      ushortx8 v0 = *(const ushortx8*)vp;
      ushortx8 v1 = *(const ushortx8*)(vp + 8);
      #pragma unroll
      for (int e = 0; e < 8; ++e) {
        vt[(dbase + e) * 32 + j]     = v0[e];
        vt[(dbase + 8 + e) * 32 + j] = v1[e];
      }
    }
    __syncthreads();
    Frag pf;
    {
      const unsigned short* pp = plds[wave] + lh * 32 + hiw * 8;
      pf.h.lo = *(const ushortx8*)pp;
      pf.h.hi = *(const ushortx8*)(pp + 16);
    }
    #pragma unroll
    for (int nt = 0; nt < 4; ++nt) {
      Frag vfr;
      const unsigned short* vp2 = vt + (nt * 16 + lh) * 32 + hiw * 8;
      vfr.h.lo = *(const ushortx8*)vp2;
      vfr.h.hi = *(const ushortx8*)(vp2 + 16);
      oo[nt] = __builtin_amdgcn_wmma_f32_16x16x32_bf16(false, pf.b, false, vfr.b, (short)0, oo[nt], false, false);
    }
    __syncthreads();
  }

  #pragma unroll
  for (int nt = 0; nt < 4; ++nt) {
    #pragma unroll
    for (int r = 0; r < 8; ++r) {
      int row = q0 + r + hiw * 8;
      if (row < SEQ) {
        float inv = 1.0f / lrun[r];
        O[base + (size_t)row * DIMC + nt * 16 + lh] = f2bf(oo[nt][r] * inv);
      }
    }
  }
}

// ---------------------------------------------------------------- launch
extern "C" void kernel_launch(void* const* d_in, const int* in_sizes, int n_in,
                              void* d_out, int out_size, void* d_ws, size_t ws_size,
                              hipStream_t stream) {
  const float* hid  = (const float*)d_in[0];
  const float* enc  = (const float*)d_in[1];
  const float* temb = (const float*)d_in[2];
  const float* rcos = (const float*)d_in[3];
  const float* rsin = (const float*)d_in[4];
  const float* w_ad1 = (const float*)d_in[5];
  const float* b_ad1 = (const float*)d_in[6];
  const float* ln1g = (const float*)d_in[7];
  const float* ln1b = (const float*)d_in[8];
  const float* wq = (const float*)d_in[9];
  const float* wk = (const float*)d_in[10];
  const float* wv = (const float*)d_in[11];
  const float* wo = (const float*)d_in[12];
  const float* bo = (const float*)d_in[13];
  const float* nqg = (const float*)d_in[14];
  const float* nqb = (const float*)d_in[15];
  const float* nkg = (const float*)d_in[16];
  const float* nkb = (const float*)d_in[17];
  const float* w_ad2 = (const float*)d_in[18];
  const float* b_ad2 = (const float*)d_in[19];
  const float* ln2g = (const float*)d_in[20];
  const float* ln2b = (const float*)d_in[21];
  const float* wff1 = (const float*)d_in[22];
  const float* bff1 = (const float*)d_in[23];
  const float* wff2 = (const float*)d_in[24];
  const float* bff2 = (const float*)d_in[25];

  float* outH = (float*)d_out;
  float* outE = outH + (size_t)BATCH * SVID * DIMC;

  char* ws = (char*)d_ws;
  size_t off = 0;
  auto alloc = [&](size_t bytes) { size_t o = off; off += (bytes + 255) & ~(size_t)255; return o; };
  float* silu_t = (float*)(ws + alloc((size_t)BATCH * TEMBC * 4));
  float* mods1  = (float*)(ws + alloc((size_t)BATCH * 6 * DIMC * 4));
  float* mods2  = (float*)(ws + alloc((size_t)BATCH * 6 * DIMC * 4));
  unsigned short* cat = (unsigned short*)(ws + alloc((size_t)MROWS * DIMC * 2));
  size_t qf_off = alloc((size_t)MROWS * DIMC * 4);
  size_t kf_off = alloc((size_t)MROWS * DIMC * 4);
  size_t vf_off = alloc((size_t)MROWS * DIMC * 4);
  size_t qb_off = alloc((size_t)MROWS * DIMC * 2);
  size_t kb_off = alloc((size_t)MROWS * DIMC * 2);
  size_t vb_off = alloc((size_t)MROWS * DIMC * 2);
  size_t ob_off = alloc((size_t)MROWS * DIMC * 2);
  float* qf = (float*)(ws + qf_off);
  float* kf = (float*)(ws + kf_off);
  float* vf = (float*)(ws + vf_off);
  unsigned short* qb16 = (unsigned short*)(ws + qb_off);
  unsigned short* kb16 = (unsigned short*)(ws + kb_off);
  unsigned short* vb16 = (unsigned short*)(ws + vb_off);
  unsigned short* ob16 = (unsigned short*)(ws + ob_off);
  // phase-2 regions overlap dead phase-1 buffers:
  float* resid1 = (float*)(ws + qf_off);                  // over qf (dead after qkrope)
  unsigned short* cat2 = (unsigned short*)(ws + ob_off);  // over ob (dead after O-proj)
  unsigned short* ffmid = (unsigned short*)(ws + vf_off); // spans vf+qb+kb (dead after attn)

  silu_k<<<(BATCH * TEMBC + 255) / 256, 256, 0, stream>>>(temb, silu_t);
  mods_k<<<(BATCH * 6 * DIMC + 255) / 256, 256, 0, stream>>>(silu_t, w_ad1, b_ad1, mods1);
  mods_k<<<(BATCH * 6 * DIMC + 255) / 256, 256, 0, stream>>>(silu_t, w_ad2, b_ad2, mods2);
  lnmod_k<<<MROWS, 256, 0, stream>>>(enc, hid, 0, ln1g, ln1b, mods1, cat);

  dim3 ggrid(DIMC / 64, (MROWS + 255) / 256);
  GemmP gq{cat, wq, nullptr, qf, nullptr, MROWS, DIMC, DIMC, nullptr, nullptr, nullptr, nullptr, nullptr};
  gemm_wmma_k<0><<<ggrid, 256, 0, stream>>>(gq);
  GemmP gk{cat, wk, nullptr, kf, nullptr, MROWS, DIMC, DIMC, nullptr, nullptr, nullptr, nullptr, nullptr};
  gemm_wmma_k<0><<<ggrid, 256, 0, stream>>>(gk);
  GemmP gv{cat, wv, nullptr, vf, nullptr, MROWS, DIMC, DIMC, nullptr, nullptr, nullptr, nullptr, nullptr};
  gemm_wmma_k<0><<<ggrid, 256, 0, stream>>>(gv);

  qkrope_k<<<(BATCH * SEQ * HEADS + 255) / 256, 256, 0, stream>>>(
      qf, kf, vf, nqg, nqb, nkg, nkb, rcos, rsin, qb16, kb16, vb16);

  attn_k<<<dim3((SEQ + 63) / 64, HEADS, BATCH), 128, 0, stream>>>(qb16, kb16, vb16, ob16);

  GemmP go{ob16, wo, bo, resid1, nullptr, MROWS, DIMC, DIMC, mods1, enc, hid, nullptr, nullptr};
  gemm_wmma_k<1><<<ggrid, 256, 0, stream>>>(go);

  lnmod_k<<<MROWS, 256, 0, stream>>>(resid1, nullptr, 1, ln2g, ln2b, mods2, cat2);

  GemmP g1{cat2, wff1, bff1, nullptr, ffmid, MROWS, FFC, DIMC, nullptr, nullptr, nullptr, nullptr, nullptr};
  gemm_wmma_k<2><<<dim3(FFC / 64, (MROWS + 255) / 256), 256, 0, stream>>>(g1);

  GemmP g2{ffmid, wff2, bff2, nullptr, nullptr, MROWS, DIMC, FFC, mods2, resid1, nullptr, outH, outE};
  gemm_wmma_k<3><<<ggrid, 256, 0, stream>>>(g2);

  (void)in_sizes; (void)n_in; (void)out_size; (void)ws_size;
}
